// PointFeaturePropagation_3865470566874
// MI455X (gfx1250) — matrix operand order repaired and checked by reading, not compiled, not run on hardware
//
#include <hip/hip_runtime.h>
#include <stdint.h>

typedef __attribute__((ext_vector_type(16))) _Float16 v16h;
typedef __attribute__((ext_vector_type(8)))  float    v8f;

static constexpr int Bb  = 4;
static constexpr int N1  = 2048;
static constexpr int N2  = 8192;
static constexpr int C1  = 256;
static constexpr int C2  = 128;
static constexpr int INC = C1 + C2;   // 384
static constexpr int OUTC = 256;
static constexpr int MTOT = Bb * N2;  // 32768

// ---------------- workspace layout (bytes) ----------------
static constexpr size_t OFF_A1   = 0;                                  // f16 [32768][384]
static constexpr size_t OFF_HID  = OFF_A1  + (size_t)MTOT * INC * 2;   // f16 [32768][256]
static constexpr size_t OFF_W1T  = OFF_HID + (size_t)MTOT * OUTC * 2;  // f16 [256][384]
static constexpr size_t OFF_W2T  = OFF_W1T + (size_t)OUTC * INC * 2;   // f16 [256][256]
static constexpr size_t OFF_SB   = OFF_W2T + (size_t)OUTC * OUTC * 2;  // f32 [4][256]

// ---------------- CDNA5 async global->LDS copy helpers ----------------
__device__ __forceinline__ void async_ld_b128(uint32_t lds_off, const _Float16* g)
{
    // GLOBAL_LOAD_ASYNC_TO_LDS_B128: VDST = LDS byte address, VADDR = 64b global
    asm volatile("global_load_async_to_lds_b128 %0, %1, off"
                 :: "v"(lds_off), "v"(g) : "memory");
}
#define S_WAIT_ASYNCCNT(N) asm volatile("s_wait_asynccnt " #N ::: "memory")

__device__ __forceinline__ uint32_t lds_addr(const void* p)
{
    // shared aperture base has zero low 32 bits -> low 32 bits = LDS offset
    return (uint32_t)(uintptr_t)p;
}

// ---------------- prep: transpose weights to f16, fold BN ----------------
__global__ __launch_bounds__(256) void prep_kernel(
    const float* __restrict__ W1, const float* __restrict__ W2,
    const float* __restrict__ b1, const float* __restrict__ g1,
    const float* __restrict__ be1, const float* __restrict__ rm1, const float* __restrict__ rv1,
    const float* __restrict__ b2, const float* __restrict__ g2,
    const float* __restrict__ be2, const float* __restrict__ rm2, const float* __restrict__ rv2,
    _Float16* __restrict__ W1t, _Float16* __restrict__ W2t, float* __restrict__ sb)
{
    int gid = blockIdx.x * blockDim.x + threadIdx.x;
    if (gid < INC * OUTC) {               // W1 [384][256] -> W1t [256][384]
        int k = gid / OUTC, n = gid % OUTC;
        W1t[(size_t)n * INC + k] = (_Float16)W1[gid];
    }
    if (gid < OUTC * OUTC) {              // W2 [256][256] -> W2t [256][256]
        int k = gid / OUTC, n = gid % OUTC;
        W2t[(size_t)n * OUTC + k] = (_Float16)W2[gid];
    }
    if (gid < OUTC) {
        float s1 = g1[gid] * rsqrtf(rv1[gid] + 1e-5f);
        sb[gid]        = s1;
        sb[256 + gid]  = (b1[gid] - rm1[gid]) * s1 + be1[gid];
        float s2 = g2[gid] * rsqrtf(rv2[gid] + 1e-5f);
        sb[512 + gid]  = s2;
        sb[768 + gid]  = (b2[gid] - rm2[gid]) * s2 + be2[gid];
    }
}

// ---------------- KNN(3) + inverse-distance interp + concat ----------------
__device__ __forceinline__ void topk_insert(float d, int i,
    float &d0, int &i0, float &d1, int &i1, float &d2, int &i2)
{
    if (d < d2 || (d == d2 && i < i2)) {
        d2 = d; i2 = i;
        if (d2 < d1 || (d2 == d1 && i2 < i1)) { float t=d1; d1=d2; d2=t; int u=i1; i1=i2; i2=u; }
        if (d1 < d0 || (d1 == d0 && i1 < i0)) { float t=d0; d0=d1; d1=t; int u=i0; i0=i1; i1=u; }
    }
}

__global__ __launch_bounds__(256) void knn_interp_kernel(
    const float* __restrict__ xyz1, const float* __restrict__ xyz2,
    const float* __restrict__ pts1, const float* __restrict__ pts2,
    _Float16* __restrict__ A1)
{
    int wave = (blockIdx.x * 256 + threadIdx.x) >> 5;   // one wave32 per query
    int lane = threadIdx.x & 31;
    int b = wave / N2;
    int q = wave - b * N2;

    const float* qp = xyz2 + ((size_t)b * N2 + q) * 3;
    float qx = qp[0], qy = qp[1], qz = qp[2];

    float d0 = 3.0e38f, d1 = 3.0e38f, d2 = 3.0e38f;
    int   i0 = 0,       i1 = 0,       i2 = 0;

    const float* pbase = xyz1 + (size_t)b * N1 * 3;
    for (int p = lane; p < N1; p += 32) {
        float dx = pbase[p * 3 + 0] - qx;
        float dy = pbase[p * 3 + 1] - qy;
        float dz = pbase[p * 3 + 2] - qz;
        float d = dx * dx + dy * dy + dz * dz;
        topk_insert(d, p, d0, i0, d1, i1, d2, i2);
    }
    // butterfly merge; all 32 lanes converge to the global top-3
    #pragma unroll
    for (int off = 16; off >= 1; off >>= 1) {
        float e0 = __shfl_xor(d0, off, 32); int j0 = __shfl_xor(i0, off, 32);
        float e1 = __shfl_xor(d1, off, 32); int j1 = __shfl_xor(i1, off, 32);
        float e2 = __shfl_xor(d2, off, 32); int j2 = __shfl_xor(i2, off, 32);
        topk_insert(e0, j0, d0, i0, d1, i1, d2, i2);
        topk_insert(e1, j1, d0, i0, d1, i1, d2, i2);
        topk_insert(e2, j2, d0, i0, d1, i1, d2, i2);
    }

    float w0 = 1.0f / (fmaxf(d0, 1e-10f) + 1e-8f);
    float w1 = 1.0f / (fmaxf(d1, 1e-10f) + 1e-8f);
    float w2 = 1.0f / (fmaxf(d2, 1e-10f) + 1e-8f);
    float ws = w0 + w1 + w2;
    w0 /= ws; w1 /= ws; w2 /= ws;

    const float* r0 = pts1 + ((size_t)b * N1 + i0) * C1;
    const float* r1 = pts1 + ((size_t)b * N1 + i1) * C1;
    const float* r2 = pts1 + ((size_t)b * N1 + i2) * C1;
    _Float16* out = A1 + (size_t)wave * INC;

    int c = lane * 8;                                   // 32 lanes * 8 = 256 ch
    #pragma unroll
    for (int j = 0; j < 8; ++j) {
        float v = w0 * r0[c + j] + w1 * r1[c + j] + w2 * r2[c + j];
        out[c + j] = (_Float16)v;
    }
    const float* p2r = pts2 + ((size_t)b * N2 + q) * C2;
    int c2 = lane * 4;                                  // 32 lanes * 4 = 128 ch
    #pragma unroll
    for (int j = 0; j < 4; ++j)
        out[C1 + c2 + j] = (_Float16)p2r[c2 + j];
}

// ---------------- WMMA GEMM + folded BN + ReLU ----------------
// C[M x 256] = A[M x K] * W[K x 256];  Bt is W transposed [256][K] in f16.
// Block tile 128x128, 8 waves (4M x 2N), each wave 2x4 WMMA 16x16 tiles.
// Double-buffered LDS filled with GLOBAL_LOAD_ASYNC_TO_LDS_B128 (ASYNCcnt).
template <int KDIM, bool OUT_F16>
__global__ __launch_bounds__(256) void gemm_bn_relu_kernel(
    const _Float16* __restrict__ A, const _Float16* __restrict__ Bt,
    const float* __restrict__ scale, const float* __restrict__ bias,
    _Float16* __restrict__ outh, float* __restrict__ outf)
{
    constexpr int KSTEPS = KDIM / 32;
    __shared__ _Float16 sA[2][128][40];   // 32 cols used, padded to 40 (80B stride)
    __shared__ _Float16 sB[2][128][40];   // [n][k]

    const int m0   = blockIdx.x * 128;
    const int n0   = blockIdx.y * 128;
    const int tid  = threadIdx.x;
    const int wave = tid >> 5;
    const int lane = tid & 31;
    const int wm   = wave & 3;          // 4 M-quadrants of 32 rows
    const int wn   = wave >> 2;         // 2 N-halves of 64 cols
    const int h    = lane >> 4;         // lane half
    const int l16  = lane & 15;

    // per-thread copy slots: 512 16B segments per matrix, 2 per thread
    const int s0 = tid,        r0s = s0 >> 2, c0s = (s0 & 3) * 8;
    const int s1 = tid + 256,  r1s = s1 >> 2, c1s = (s1 & 3) * 8;
    const _Float16* gA0 = A  + (size_t)(m0 + r0s) * KDIM + c0s;
    const _Float16* gA1 = A  + (size_t)(m0 + r1s) * KDIM + c1s;
    const _Float16* gB0 = Bt + (size_t)(n0 + r0s) * KDIM + c0s;
    const _Float16* gB1 = Bt + (size_t)(n0 + r1s) * KDIM + c1s;

    uint32_t lA0[2], lA1[2], lB0[2], lB1[2];
    #pragma unroll
    for (int bi = 0; bi < 2; ++bi) {
        lA0[bi] = lds_addr(&sA[bi][r0s][c0s]);
        lA1[bi] = lds_addr(&sA[bi][r1s][c1s]);
        lB0[bi] = lds_addr(&sB[bi][r0s][c0s]);
        lB1[bi] = lds_addr(&sB[bi][r1s][c1s]);
    }

    v8f c[2][4];
    #pragma unroll
    for (int mi = 0; mi < 2; ++mi)
        #pragma unroll
        for (int ni = 0; ni < 4; ++ni)
            c[mi][ni] = (v8f){0.f, 0.f, 0.f, 0.f, 0.f, 0.f, 0.f, 0.f};

    // prologue: async-fetch tile 0 into buffer 0
    async_ld_b128(lA0[0], gA0);
    async_ld_b128(lA1[0], gA1);
    async_ld_b128(lB0[0], gB0);
    async_ld_b128(lB1[0], gB1);

    for (int ks = 0; ks < KSTEPS; ++ks) {
        const int cur = ks & 1;
        if (ks + 1 < KSTEPS) {
            // prefetch next tile into the other buffer (overlaps with WMMAs)
            const int nxt = cur ^ 1;
            const int ko  = (ks + 1) * 32;
            async_ld_b128(lA0[nxt], gA0 + ko);
            async_ld_b128(lA1[nxt], gA1 + ko);
            async_ld_b128(lB0[nxt], gB0 + ko);
            async_ld_b128(lB1[nxt], gB1 + ko);
            S_WAIT_ASYNCCNT(4);   // my 4 oldest (current tile) have landed
        } else {
            S_WAIT_ASYNCCNT(0);
        }
        __syncthreads();          // everyone's current-tile data visible

        // A fragments: ISA layout — e<8: K=8h+e ; e>=8: K=16+8h+(e-8)
        v16h afr[2];
        #pragma unroll
        for (int mi = 0; mi < 2; ++mi) {
            int row = wm * 32 + mi * 16 + l16;
            union { v16h v; uint32_t u[8]; } ua;
            const uint32_t* rp = (const uint32_t*)&sA[cur][row][0];
            #pragma unroll
            for (int w = 0; w < 8; ++w) {
                int kk = (w < 4) ? (8 * h + 2 * w) : (16 + 8 * h + 2 * (w - 4));
                ua.u[w] = rp[kk >> 1];
            }
            afr[mi] = ua.v;
        }
        // B fragments: e<8: K=8h+e ; e>=8: K=16+8h+(e-8), column = l16 (Bt stored [n][k])
        v16h bfr[4];
        #pragma unroll
        for (int ni = 0; ni < 4; ++ni) {
            int col = wn * 64 + ni * 16 + l16;
            union { v16h v; uint32_t u[8]; } ub;
            const uint32_t* rp = (const uint32_t*)&sB[cur][col][0];
            #pragma unroll
            for (int w = 0; w < 8; ++w)
                ub.u[w] = rp[((w < 4) ? (8 * h + 2 * w) : (16 + 8 * h + 2 * (w - 4))) >> 1];
            bfr[ni] = ub.v;
        }

        #pragma unroll
        for (int mi = 0; mi < 2; ++mi)
            #pragma unroll
            for (int ni = 0; ni < 4; ++ni)
                c[mi][ni] = __builtin_amdgcn_wmma_f32_16x16x32_f16(
                    false, afr[mi], false, bfr[ni],
                    (short)0, c[mi][ni], false, false);

        __syncthreads();          // protect buffer (cur gets overwritten next iter)
    }

    // epilogue: y = relu(acc * scale[n] + bias[n]); D layout: VGPR r -> M = 8h + r
    #pragma unroll
    for (int ni = 0; ni < 4; ++ni) {
        int col = n0 + wn * 64 + ni * 16 + l16;
        float s  = scale[col];
        float bb = bias[col];
        #pragma unroll
        for (int mi = 0; mi < 2; ++mi) {
            int rowbase = m0 + wm * 32 + mi * 16 + h * 8;
            #pragma unroll
            for (int r = 0; r < 8; ++r) {
                float v = fmaxf(c[mi][ni][r] * s + bb, 0.0f);
                if (OUT_F16) outh[(size_t)(rowbase + r) * OUTC + col] = (_Float16)v;
                else         outf[(size_t)(rowbase + r) * OUTC + col] = v;
            }
        }
    }
}

// ---------------- launch ----------------
extern "C" void kernel_launch(void* const* d_in, const int* in_sizes, int n_in,
                              void* d_out, int out_size, void* d_ws, size_t ws_size,
                              hipStream_t stream)
{
    (void)in_sizes; (void)n_in; (void)out_size; (void)ws_size;

    const float* xyz1 = (const float*)d_in[0];
    const float* xyz2 = (const float*)d_in[1];
    const float* pts1 = (const float*)d_in[2];
    const float* pts2 = (const float*)d_in[3];
    const float* W1   = (const float*)d_in[4];
    const float* b1   = (const float*)d_in[5];
    const float* g1   = (const float*)d_in[6];
    const float* be1  = (const float*)d_in[7];
    const float* rm1  = (const float*)d_in[8];
    const float* rv1  = (const float*)d_in[9];
    const float* W2   = (const float*)d_in[10];
    const float* b2   = (const float*)d_in[11];
    const float* g2   = (const float*)d_in[12];
    const float* be2  = (const float*)d_in[13];
    const float* rm2  = (const float*)d_in[14];
    const float* rv2  = (const float*)d_in[15];

    char* ws = (char*)d_ws;
    _Float16* A1     = (_Float16*)(ws + OFF_A1);
    _Float16* hidden = (_Float16*)(ws + OFF_HID);
    _Float16* W1t    = (_Float16*)(ws + OFF_W1T);
    _Float16* W2t    = (_Float16*)(ws + OFF_W2T);
    float*    sb     = (float*)   (ws + OFF_SB);
    float*    out    = (float*)d_out;

    // 1) fold BN, transpose+convert weights (98304 threads covers 384*256)
    prep_kernel<<<(INC * OUTC) / 256, 256, 0, stream>>>(
        W1, W2, b1, g1, be1, rm1, rv1, b2, g2, be2, rm2, rv2, W1t, W2t, sb);

    // 2) KNN + interpolation + concat -> f16 activation matrix [32768 x 384]
    knn_interp_kernel<<<MTOT / 8, 256, 0, stream>>>(xyz1, xyz2, pts1, pts2, A1);

    // 3) layer 1: [32768 x 384] x [384 x 256] -> f16 hidden
    gemm_bn_relu_kernel<INC, true><<<dim3(MTOT / 128, OUTC / 128), 256, 0, stream>>>(
        A1, W1t, sb, sb + 256, hidden, nullptr);

    // 4) layer 2: [32768 x 256] x [256 x 256] -> f32 output
    gemm_bn_relu_kernel<OUTC, false><<<dim3(MTOT / 128, OUTC / 128), 256, 0, stream>>>(
        hidden, W2t, sb + 512, sb + 768, nullptr, out);
}
